// ProteinAttention_89979564851292
// MI455X (gfx1250) — compile-verified, matrix operand order checked by reading, not run on hardware
//
#include <hip/hip_runtime.h>
#include <hip/hip_bf16.h>

// Problem constants (from reference)
#define BB 8
#define NN 2048
#define CC 512
#define HH 8
#define DD 64
#define SCALE_Q 0.125f   // HEAD_DIM^-0.5 = 1/8

typedef __bf16 bf16;
typedef __attribute__((ext_vector_type(16))) __bf16 v16bf;
typedef __attribute__((ext_vector_type(8)))  float  v8f;
typedef __attribute__((ext_vector_type(4)))  unsigned int u32x4;
typedef __attribute__((ext_vector_type(8)))  int i32x8;
typedef __attribute__((ext_vector_type(4)))  int i32x4;

union FragU {
    v16bf v;
    unsigned int u[8];
};
union PackU {
    unsigned int u;
    bf16 h[2];
};

__device__ static inline v8f zero8() {
    v8f z;
#pragma unroll
    for (int i = 0; i < 8; ++i) z[i] = 0.0f;
    return z;
}

__device__ static inline v8f wmma_bf16(v16bf a, v16bf b, v8f c) {
    return __builtin_amdgcn_wmma_f32_16x16x32_bf16(false, a, false, b, (short)0, c,
                                                   false, false);
}

// K index pattern for 16-bit A-matrix 16x32 fragment (CDNA5 ISA 7.12.2)
__device__ static inline int kpat(int v, int half) {
    int base = (v < 4) ? (v * 2) : (16 + (v - 4) * 2);
    return base + half * 8;
}

// 16x32 A-style fragment from row-major bf16 (packed 32-bit reads only).
// Also serves as B fragment whenever B^T is row-major.
__device__ static inline v16bf load_fragA_rm(const bf16* p, int ld) {
    int lane = threadIdx.x & 31;
    int m = lane & 15, half = lane >> 4;
    const bf16* row = p + m * ld;
    FragU f;
#pragma unroll
    for (int v = 0; v < 8; ++v) {
        int k = kpat(v, half);
        f.u[v] = *(const unsigned int*)(row + k);
    }
    return f.v;
}

// ---------------------------------------------------------------------------
// Tensor Data Mover: 2D bf16 tile global -> LDS with per-row padding.
// pad_code: pads 1 DWORD (2 bf16) after every (2<<pad_code) DWORDs:
//   tile row 32 elem (16 dw) -> code 3 => LDS row stride 34 elems
//   tile row 64 elem (32 dw) -> code 4 => LDS row stride 66 elems
// ---------------------------------------------------------------------------
#if defined(__has_builtin)
#if __has_builtin(__builtin_amdgcn_tensor_load_to_lds)
#define HAVE_TDM 1
#endif
#endif

#ifdef HAVE_TDM
__device__ static inline void tdm_load_tile(unsigned lds_off, const void* gptr,
                                            unsigned tile_w, unsigned tile_h,
                                            unsigned row_stride, unsigned pad_code) {
    unsigned long long ga = (unsigned long long)(uintptr_t)gptr;
    u32x4 g0;
    g0[0] = 1u;                                             // count=1, user descriptor
    g0[1] = lds_off;                                        // lds_addr (bytes)
    g0[2] = (unsigned)ga;                                   // global_addr[31:0]
    g0[3] = (unsigned)((ga >> 32) & 0x1FFFFFFull) | (2u << 30);  // addr[56:32] | type=2
    i32x8 g1;
    unsigned w0 = (1u << 16)                                // data_size=1 (2 bytes)
                | (1u << 20)                                // pad_enable
                | (pad_code << 22);                         // pad_interval (pad_amount=0 -> 1 dw)
    g1[0] = (int)w0;
    g1[1] = (int)((row_stride & 0xFFFFu) << 16);            // tensor_dim0[15:0]  @ [63:48]
    g1[2] = (int)(((row_stride >> 16) & 0xFFFFu)            // tensor_dim0[31:16]
                | ((tile_h & 0xFFFFu) << 16));              // tensor_dim1[15:0]
    g1[3] = (int)(((tile_h >> 16) & 0xFFFFu)                // tensor_dim1[31:16]
                | ((tile_w & 0xFFFFu) << 16));              // tile_dim0
    g1[4] = (int)(tile_h & 0xFFFFu);                        // tile_dim1 (tile_dim2=0)
    g1[5] = (int)row_stride;                                // tensor_dim0_stride[31:0]
    g1[6] = 0;
    g1[7] = 0;
    i32x4 z4 = {0, 0, 0, 0};
#if __clang_major__ >= 23
    i32x8 z8 = {0, 0, 0, 0, 0, 0, 0, 0};
    __builtin_amdgcn_tensor_load_to_lds(g0, g1, z4, z4, z8, 0);
#else
    __builtin_amdgcn_tensor_load_to_lds(g0, g1, z4, z4, 0);
#endif
}
#endif

// ---------------------------------------------------------------------------
// Kernel 0: fp32 -> bf16 weight conversion, TRANSPOSED (w^T row-major) so all
// downstream B fragments are packed-32-bit loads.
// ---------------------------------------------------------------------------
__global__ void convert_weights(const float* __restrict__ wq, const float* __restrict__ wp,
                                bf16* __restrict__ wq_t, bf16* __restrict__ wp_t) {
    int i = blockIdx.x * 256 + threadIdx.x;
    if (i < 3 * CC * CC) {                       // wq_t[n][k], n<1536, k<512
        int n = i >> 9, k = i & 511;
        wq_t[i] = (bf16)wq[(size_t)k * (3 * CC) + n];
    }
    if (i < CC * CC) {                           // wp_t[n][k], n<512, k<512
        int n = i >> 9, k = i & 511;
        wp_t[i] = (bf16)wp[(size_t)k * CC + n];
    }
}

// ---------------------------------------------------------------------------
// Kernel 1: QKV projection GEMM (16384x512)@(512x1536), bf16 WMMA.
// Double-buffered: TDM B^T tile and register-staged A tile for iteration i+1
// overlap the WMMA block of iteration i.  One barrier per K step.
// ---------------------------------------------------------------------------
#define GBM 128
#define GBN 128
#define GBK 32
#define NIT (CC / GBK)

__global__ __launch_bounds__(256) void qkv_gemm(const float* __restrict__ x,
                                                const bf16* __restrict__ wq_t,
                                                bf16* __restrict__ qb,
                                                bf16* __restrict__ kb,
                                                bf16* __restrict__ vt) {
    __shared__ bf16 As[2][GBM][GBK + 2];    // stride 34
    __shared__ bf16 BsT[2][GBN][GBK + 2];   // stride 34 (matches TDM pad)

    const int t = threadIdx.x;
    const int wave = t >> 5;
    const int wm = wave & 3;
    const int wn = wave >> 2;
    const int m0 = blockIdx.x * GBM;
    const int n0 = blockIdx.y * GBN;

    v8f acc[2][4];
#pragma unroll
    for (int i = 0; i < 2; ++i)
#pragma unroll
        for (int j = 0; j < 4; ++j) acc[i][j] = zero8();

    // ---- prologue: fill buffer 0 ----
#pragma unroll
    for (int i = 0; i < 8; ++i) {
        int e = t + i * 256;                 // 2048 pairs
        int r = e >> 4, c = (e & 15) * 2;
        float2 a2 = *(const float2*)&x[(size_t)(m0 + r) * CC + c];
        PackU pk;
        pk.h[0] = (bf16)a2.x;
        pk.h[1] = (bf16)a2.y;
        *(unsigned*)&As[0][r][c] = pk.u;
    }
#ifdef HAVE_TDM
    if (wave == 0) {
        tdm_load_tile((unsigned)(uintptr_t)&BsT[0][0][0],
                      wq_t + (size_t)n0 * CC, GBK, GBN, CC, 3);
        __builtin_amdgcn_s_wait_tensorcnt(0);
    }
#else
#pragma unroll
    for (int i = 0; i < 16; ++i) {
        int e = t + i * 256;
        int r = e >> 5, c = e & 31;
        BsT[0][r][c] = wq_t[(size_t)(n0 + r) * CC + c];
    }
#endif
    __syncthreads();

    for (int it = 0; it < NIT; ++it) {
        const int cur = it & 1, nxt = cur ^ 1;
        const int kn = (it + 1) * GBK;
        const bool has_next = (it + 1 < NIT);

        // ---- stage next A tile into registers; kick next TDM ----
        float2 av[8];
        if (has_next) {
#pragma unroll
            for (int i = 0; i < 8; ++i) {
                int e = t + i * 256;
                int r = e >> 4, c = (e & 15) * 2;
                av[i] = *(const float2*)&x[(size_t)(m0 + r) * CC + kn + c];
            }
#ifdef HAVE_TDM
            if (wave == 0)
                tdm_load_tile((unsigned)(uintptr_t)&BsT[nxt][0][0],
                              wq_t + (size_t)n0 * CC + kn, GBK, GBN, CC, 3);
#endif
        }

        // ---- compute current tile ----
        v16bf af[2], bfg[4];
#pragma unroll
        for (int mt = 0; mt < 2; ++mt)
            af[mt] = load_fragA_rm(&As[cur][wm * 32 + mt * 16][0], GBK + 2);
#pragma unroll
        for (int nt = 0; nt < 4; ++nt)
            bfg[nt] = load_fragA_rm(&BsT[cur][wn * 64 + nt * 16][0], GBK + 2);
#pragma unroll
        for (int mt = 0; mt < 2; ++mt)
#pragma unroll
            for (int nt = 0; nt < 4; ++nt)
                acc[mt][nt] = wmma_bf16(af[mt], bfg[nt], acc[mt][nt]);

        // ---- commit next A tile; drain TDM; single barrier ----
        if (has_next) {
#pragma unroll
            for (int i = 0; i < 8; ++i) {
                int e = t + i * 256;
                int r = e >> 4, c = (e & 15) * 2;
                PackU pk;
                pk.h[0] = (bf16)av[i].x;
                pk.h[1] = (bf16)av[i].y;
                *(unsigned*)&As[nxt][r][c] = pk.u;
            }
#ifdef HAVE_TDM
            if (wave == 0) __builtin_amdgcn_s_wait_tensorcnt(0);
#else
#pragma unroll
            for (int i = 0; i < 16; ++i) {
                int e = t + i * 256;
                int r = e >> 5, c = e & 31;
                BsT[nxt][r][c] = wq_t[(size_t)(n0 + r) * CC + kn + c];
            }
#endif
        }
        __syncthreads();
    }

    // ---- epilogue: tile-uniform selectors, scalar branches only ----
    const int lane = t & 31;
    const int lm = (lane >> 4) * 8, ln = lane & 15;
    const int three = n0 >> 9;
    const int bidx = m0 >> 11;
    const int nbase = (m0 & 2047) + wm * 32;
#pragma unroll
    for (int mt = 0; mt < 2; ++mt) {
#pragma unroll
        for (int nt = 0; nt < 4; ++nt) {
            int jb = n0 + wn * 64 + nt * 16;
            int h = (jb >> 6) & 7;
            int dbase = jb & 63;
            if (three == 0) {
                bf16* dst = qb + ((size_t)(bidx * HH + h)) * NN * DD;
#pragma unroll
                for (int r = 0; r < 8; ++r) {
                    int n = nbase + mt * 16 + lm + r;
                    dst[(size_t)n * DD + dbase + ln] = (bf16)(acc[mt][nt][r] * SCALE_Q);
                }
            } else if (three == 1) {
                bf16* dst = kb + ((size_t)(bidx * HH + h)) * NN * DD;
#pragma unroll
                for (int r = 0; r < 8; ++r) {
                    int n = nbase + mt * 16 + lm + r;
                    dst[(size_t)n * DD + dbase + ln] = (bf16)acc[mt][nt][r];
                }
            } else {                     // V stored TRANSPOSED: vt[b,h,d,n]
                bf16* dst = vt + ((size_t)(bidx * HH + h)) * DD * NN;
#pragma unroll
                for (int r = 0; r < 8; ++r) {
                    int n = nbase + mt * 16 + lm + r;
                    dst[(size_t)(dbase + ln) * NN + n] = (bf16)acc[mt][nt][r];
                }
            }
        }
    }
}

// ---------------------------------------------------------------------------
// Kernel 2: flash attention per (b,h).  K / V^T tiles double-buffered via TDM;
// the DMA for key-block i+1 overlaps the whole softmax+WMMA of block i.
// ---------------------------------------------------------------------------
__global__ __launch_bounds__(256) void flash_attn(const bf16* __restrict__ qb,
                                                  const bf16* __restrict__ kb,
                                                  const bf16* __restrict__ vt,
                                                  const unsigned char* __restrict__ mask,
                                                  bf16* __restrict__ ob) {
    __shared__ bf16 Ks[2][64][66];       // keys x d     (stride 66 = TDM pad)
    __shared__ bf16 Vts[2][64][66];      // d    x keys
    __shared__ bf16 Ps[8][16 * 66];      // per-wave P bounce

    const int t = threadIdx.x;
    const int wave = t >> 5;
    const int lane = t & 31;
    const int bh = blockIdx.y;
    const int bidx = bh >> 3;
    const int h = bh & 7;
    const int q0 = blockIdx.x * 128 + wave * 16;

    const bf16* Q = qb + ((size_t)bh * NN + q0) * DD;
    const bf16* K = kb + (size_t)bh * NN * DD;
    const bf16* Vt = vt + (size_t)bh * DD * NN;

    v16bf qf0 = load_fragA_rm(Q, DD);
    v16bf qf1 = load_fragA_rm(Q + 32, DD);

    v8f o[4];
#pragma unroll
    for (int i = 0; i < 4; ++i) o[i] = zero8();
    float mrow[8], lrow[8];
#pragma unroll
    for (int r = 0; r < 8; ++r) { mrow[r] = -1e30f; lrow[r] = 0.0f; }

    const int lm = (lane >> 4) * 8, ln = lane & 15;
    bf16* P = &Ps[wave][0];

    // ---- prologue: key block 0 into buffer 0 ----
#ifdef HAVE_TDM
    if (wave == 0) {
        tdm_load_tile((unsigned)(uintptr_t)&Ks[0][0][0],  K,  64, 64, DD, 4);
        tdm_load_tile((unsigned)(uintptr_t)&Vts[0][0][0], Vt, 64, 64, NN, 4);
        __builtin_amdgcn_s_wait_tensorcnt(0);
    }
#else
#pragma unroll
    for (int i = 0; i < 16; ++i) {
        int e = t + i * 256;
        int r = e >> 6, c = e & 63;
        Ks[0][r][c]  = K[(size_t)r * DD + c];
        Vts[0][r][c] = Vt[(size_t)r * NN + c];
    }
#endif
    __syncthreads();

    const int NB = NN / 64;              // 32 key blocks
    for (int ib = 0; ib < NB; ++ib) {
        const int kb0 = ib * 64;
        const int cur = ib & 1, nxt = cur ^ 1;
        const bool has_next = (ib + 1 < NB);

#ifdef HAVE_TDM
        if (has_next && wave == 0) {
            tdm_load_tile((unsigned)(uintptr_t)&Ks[nxt][0][0],
                          K + (size_t)(kb0 + 64) * DD, 64, 64, DD, 4);
            tdm_load_tile((unsigned)(uintptr_t)&Vts[nxt][0][0],
                          Vt + kb0 + 64, 64, 64, NN, 4);
        }
#endif
        // ---- S = Q @ K^T ----
        v8f s[4];
#pragma unroll
        for (int nt = 0; nt < 4; ++nt) {
            v16bf kf0 = load_fragA_rm(&Ks[cur][nt * 16][0], 66);
            v16bf kf1 = load_fragA_rm(&Ks[cur][nt * 16][32], 66);
            v8f a = zero8();
            a = wmma_bf16(qf0, kf0, a);
            a = wmma_bf16(qf1, kf1, a);
            s[nt] = a;
        }
        // ---- key mask ----
#pragma unroll
        for (int nt = 0; nt < 4; ++nt) {
            int jk = kb0 + nt * 16 + ln;
            if (!mask[bidx * NN + jk]) {
#pragma unroll
                for (int r = 0; r < 8; ++r) s[nt][r] = -1e30f;
            }
        }
        // ---- online softmax ----
#pragma unroll
        for (int r = 0; r < 8; ++r) {
            float rm = fmaxf(fmaxf(s[0][r], s[1][r]), fmaxf(s[2][r], s[3][r]));
            rm = fmaxf(rm, __shfl_xor(rm, 1, 32));
            rm = fmaxf(rm, __shfl_xor(rm, 2, 32));
            rm = fmaxf(rm, __shfl_xor(rm, 4, 32));
            rm = fmaxf(rm, __shfl_xor(rm, 8, 32));
            float mnew = fmaxf(mrow[r], rm);
            float alpha = __expf(mrow[r] - mnew);
            float rs = 0.0f;
#pragma unroll
            for (int nt = 0; nt < 4; ++nt) {
                float p = __expf(s[nt][r] - mnew);
                s[nt][r] = p;
                rs += p;
            }
            rs += __shfl_xor(rs, 1, 32);
            rs += __shfl_xor(rs, 2, 32);
            rs += __shfl_xor(rs, 4, 32);
            rs += __shfl_xor(rs, 8, 32);
            lrow[r] = lrow[r] * alpha + rs;
            mrow[r] = mnew;
#pragma unroll
            for (int nt = 0; nt < 4; ++nt) o[nt][r] *= alpha;
        }
        // ---- P: C/D layout -> row-major bf16 in per-wave LDS ----
#pragma unroll
        for (int nt = 0; nt < 4; ++nt)
#pragma unroll
            for (int r = 0; r < 8; ++r)
                P[(lm + r) * 66 + nt * 16 + ln] = (bf16)s[nt][r];
        // ---- O += P @ V ----
#pragma unroll
        for (int kc = 0; kc < 2; ++kc) {
            v16bf pf = load_fragA_rm(P + kc * 32, 66);
#pragma unroll
            for (int nt = 0; nt < 4; ++nt) {
                v16bf vf = load_fragA_rm(&Vts[cur][nt * 16][kc * 32], 66);
                o[nt] = wmma_bf16(pf, vf, o[nt]);
            }
        }

        if (has_next) {
#ifdef HAVE_TDM
            if (wave == 0) __builtin_amdgcn_s_wait_tensorcnt(0);
#else
#pragma unroll
            for (int i = 0; i < 16; ++i) {
                int e = t + i * 256;
                int r = e >> 6, c = e & 63;
                Ks[nxt][r][c]  = K[(size_t)(kb0 + 64 + r) * DD + c];
                Vts[nxt][r][c] = Vt[(size_t)r * NN + kb0 + 64 + c];
            }
#endif
        }
        __syncthreads();
    }

    // ---- epilogue: O /= l via rcp, write [B, N, H*D] bf16 ----
#pragma unroll
    for (int r = 0; r < 8; ++r) {
        float inv = __builtin_amdgcn_rcpf(lrow[r]);
        int row = q0 + lm + r;
#pragma unroll
        for (int nt = 0; nt < 4; ++nt) {
            int col = h * DD + nt * 16 + ln;
            ob[((size_t)bidx * NN + row) * CC + col] = (bf16)(o[nt][r] * inv);
        }
    }
}

// ---------------------------------------------------------------------------
// Kernel 3: output projection (16384x512)@(512x512)+bias -> fp32.
// Both tiles double-buffered through the TDM.
// ---------------------------------------------------------------------------
__global__ __launch_bounds__(256) void out_proj(const bf16* __restrict__ a,
                                                const bf16* __restrict__ wp_t,
                                                const float* __restrict__ bias,
                                                float* __restrict__ out) {
    __shared__ bf16 As[2][GBM][GBK + 2];
    __shared__ bf16 BsT[2][GBN][GBK + 2];

    const int t = threadIdx.x;
    const int wave = t >> 5;
    const int wm = wave & 3;
    const int wn = wave >> 2;
    const int m0 = blockIdx.x * GBM;
    const int n0 = blockIdx.y * GBN;

    v8f acc[2][4];
#pragma unroll
    for (int i = 0; i < 2; ++i)
#pragma unroll
        for (int j = 0; j < 4; ++j) acc[i][j] = zero8();

#ifdef HAVE_TDM
    if (wave == 0) {
        tdm_load_tile((unsigned)(uintptr_t)&As[0][0][0],  a + (size_t)m0 * CC,    GBK, GBM, CC, 3);
        tdm_load_tile((unsigned)(uintptr_t)&BsT[0][0][0], wp_t + (size_t)n0 * CC, GBK, GBN, CC, 3);
        __builtin_amdgcn_s_wait_tensorcnt(0);
    }
#else
#pragma unroll
    for (int i = 0; i < 16; ++i) {
        int e = t + i * 256;
        int r = e >> 5, c = e & 31;
        As[0][r][c]  = a[(size_t)(m0 + r) * CC + c];
        BsT[0][r][c] = wp_t[(size_t)(n0 + r) * CC + c];
    }
#endif
    __syncthreads();

    for (int it = 0; it < NIT; ++it) {
        const int cur = it & 1, nxt = cur ^ 1;
        const int kn = (it + 1) * GBK;
        const bool has_next = (it + 1 < NIT);

#ifdef HAVE_TDM
        if (has_next && wave == 0) {
            tdm_load_tile((unsigned)(uintptr_t)&As[nxt][0][0],
                          a + (size_t)m0 * CC + kn, GBK, GBM, CC, 3);
            tdm_load_tile((unsigned)(uintptr_t)&BsT[nxt][0][0],
                          wp_t + (size_t)n0 * CC + kn, GBK, GBN, CC, 3);
        }
#endif
        v16bf af[2], bfg[4];
#pragma unroll
        for (int mt = 0; mt < 2; ++mt)
            af[mt] = load_fragA_rm(&As[cur][wm * 32 + mt * 16][0], GBK + 2);
#pragma unroll
        for (int nt = 0; nt < 4; ++nt)
            bfg[nt] = load_fragA_rm(&BsT[cur][wn * 64 + nt * 16][0], GBK + 2);
#pragma unroll
        for (int mt = 0; mt < 2; ++mt)
#pragma unroll
            for (int nt = 0; nt < 4; ++nt)
                acc[mt][nt] = wmma_bf16(af[mt], bfg[nt], acc[mt][nt]);

        if (has_next) {
#ifdef HAVE_TDM
            if (wave == 0) __builtin_amdgcn_s_wait_tensorcnt(0);
#else
#pragma unroll
            for (int i = 0; i < 16; ++i) {
                int e = t + i * 256;
                int r = e >> 5, c = e & 31;
                As[nxt][r][c]  = a[(size_t)(m0 + r) * CC + kn + c];
                BsT[nxt][r][c] = wp_t[(size_t)(n0 + r) * CC + kn + c];
            }
#endif
        }
        __syncthreads();
    }

    const int lane = t & 31;
    const int lm = (lane >> 4) * 8, ln = lane & 15;
#pragma unroll
    for (int mt = 0; mt < 2; ++mt) {
#pragma unroll
        for (int nt = 0; nt < 4; ++nt) {
#pragma unroll
            for (int r = 0; r < 8; ++r) {
                int gi = m0 + wm * 32 + mt * 16 + lm + r;
                int gj = n0 + wn * 64 + nt * 16 + ln;
                out[(size_t)gi * CC + gj] = acc[mt][nt][r] + bias[gj];
            }
        }
    }
}

// ---------------------------------------------------------------------------
extern "C" void kernel_launch(void* const* d_in, const int* in_sizes, int n_in,
                              void* d_out, int out_size, void* d_ws, size_t ws_size,
                              hipStream_t stream) {
    const float*         x      = (const float*)d_in[0];
    const unsigned char* mask   = (const unsigned char*)d_in[1];   // jnp.bool_ -> 1B
    const float*         w_qkv  = (const float*)d_in[2];
    const float*         w_proj = (const float*)d_in[3];
    const float*         b_proj = (const float*)d_in[4];
    float*               out    = (float*)d_out;

    bf16* wq_t = (bf16*)d_ws;                               // 1536*512
    bf16* wp_t = wq_t + (size_t)3 * CC * CC;                // 512*512
    bf16* qb   = wp_t + (size_t)CC * CC;                    // B*H*N*D
    bf16* kbuf = qb + (size_t)BB * HH * NN * DD;
    bf16* vt   = kbuf + (size_t)BB * HH * NN * DD;          // transposed V
    bf16* ob   = vt + (size_t)BB * HH * NN * DD;            // B*N*C
    // ~69 MB total

    convert_weights<<<(3 * CC * CC + 255) / 256, 256, 0, stream>>>(w_qkv, w_proj, wq_t, wp_t);

    qkv_gemm<<<dim3(16384 / GBM, (3 * CC) / GBN), 256, 0, stream>>>(x, wq_t, qb, kbuf, vt);

    flash_attn<<<dim3(NN / 128, BB * HH), 256, 0, stream>>>(qb, kbuf, vt, mask, ob);

    out_proj<<<dim3(16384 / GBM, CC / GBN), 256, 0, stream>>>(ob, wp_t, b_proj, out);
}
